// MainModel_75514114998350
// MI455X (gfx1250) — matrix operand order repaired and checked by reading, not compiled, not run on hardware
//
#include <hip/hip_runtime.h>

// ---------------------------------------------------------------------------
// Types for CDNA5 WMMA (wave32, 16x16x32 bf16 -> f32)
// ---------------------------------------------------------------------------
typedef __attribute__((ext_vector_type(16))) __bf16 bf16x16;
typedef __attribute__((ext_vector_type(8)))  __bf16 bf16x8;
typedef __attribute__((ext_vector_type(8)))  float  f32x8;

#define DEV __device__ __forceinline__

// Problem constants (from reference)
constexpr int Bb = 4, Ss = 2048, Ee = 512, Hh = 8, Dd = 64;
constexpr int MROWS = Bb * Ss;      // 8192 token rows
constexpr int QKVN  = 3 * Ee;       // 1536
constexpr int FFN   = 2304;
constexpr int LDSTR = 80;           // LDS row stride in bytes (64B data + 16B pad)

DEV bf16x16 cat16(bf16x8 lo, bf16x8 hi) {
  return __builtin_shufflevector(lo, hi, 0,1,2,3,4,5,6,7,8,9,10,11,12,13,14,15);
}
DEV f32x8 wmma_bf16(bf16x16 a, bf16x16 b, f32x8 c) {
  // (neg_a, A, neg_b, B, c_mod, C, reuse_a, reuse_b)
  return __builtin_amdgcn_wmma_f32_16x16x32_bf16(false, a, false, b, (short)0, c,
                                                 false, false);
}

// CDNA5 async global->LDS copy (ASYNCcnt tracked). LDS address = low 32 bits
// of the generic address.
DEV void async_load_b128(void* lds_ptr, const void* gptr) {
  unsigned lds_off = (unsigned)(unsigned long long)lds_ptr;
  unsigned long long ga = (unsigned long long)gptr;
  asm volatile("global_load_async_to_lds_b128 %0, %1, off"
               :: "v"(lds_off), "v"(ga)
               : "memory");
}
DEV void wait_async0() {
  asm volatile("s_wait_asynccnt 0x0" ::: "memory");
}

DEV float block_sum(float v, float* red) {
  const int t = threadIdx.x;
  red[t] = v;
  __syncthreads();
  for (int off = 128; off > 0; off >>= 1) {
    if (t < off) red[t] += red[t + off];
    __syncthreads();
  }
  float r = red[0];
  __syncthreads();
  return r;
}

// ---------------------------------------------------------------------------
// Small utility kernels
// ---------------------------------------------------------------------------
__global__ void zero_f_kernel(float* p, int n) {
  int i = blockIdx.x * 256 + threadIdx.x;
  if (i < n) p[i] = 0.0f;
}

__global__ void cast_bf_kernel(const float* __restrict__ in, __bf16* __restrict__ out, int n) {
  int i = blockIdx.x * 256 + threadIdx.x;
  if (i < n) out[i] = (__bf16)in[i];
}

__global__ void finaladd_kernel(const float* __restrict__ a, const float* __restrict__ b,
                                float* __restrict__ o, int n) {
  int i = blockIdx.x * 256 + threadIdx.x;
  if (i < n) o[i] = a[i] + b[i];
}

// ---------------------------------------------------------------------------
// 1) RMSNorm(x, rms1_w) -> h (bf16) + per-batch sum/sumsq of h + column sums
// ---------------------------------------------------------------------------
__global__ __launch_bounds__(256) void rmsnorm1_kernel(
    const float* __restrict__ x, const float* __restrict__ w,
    __bf16* __restrict__ h, float* __restrict__ bstat, float* __restrict__ xm) {
  __shared__ float red[256];
  const int tid = threadIdx.x;
  const int row = blockIdx.x;         // b*S + s
  const int b = row >> 11;            // S == 2048
  const float* xr = x + (size_t)row * Ee;
  float v0 = xr[tid], v1 = xr[tid + 256];
  float total = block_sum(v0 * v0 + v1 * v1, red);
  float rms = rsqrtf(total * (1.0f / (float)Ee) + 1e-6f);
  float h0 = v0 * rms * w[tid];
  float h1 = v1 * rms * w[tid + 256];
  __bf16* hr = h + (size_t)row * Ee;
  hr[tid] = (__bf16)h0;
  hr[tid + 256] = (__bf16)h1;
  float s  = block_sum(h0 + h1, red);
  float sq = block_sum(h0 * h0 + h1 * h1, red);
  if (tid == 0) {
    atomicAdd(&bstat[2 * b + 0], s);
    atomicAdd(&bstat[2 * b + 1], sq);
  }
  atomicAdd(&xm[b * Ee + tid], h0);
  atomicAdd(&xm[b * Ee + tid + 256], h1);
}

// ---------------------------------------------------------------------------
// 2) complexity -> dynamic window size (single tiny block)
// ---------------------------------------------------------------------------
__global__ __launch_bounds__(128) void cswin_kernel(
    const float* __restrict__ bstat, const float* __restrict__ xm,
    const float* __restrict__ w1, const float* __restrict__ w2, int* __restrict__ wsout) {
  __shared__ float red[128];
  __shared__ float accs;
  const int t = threadIdx.x;
  if (t == 0) accs = 0.0f;
  __syncthreads();
  for (int b = 0; b < Bb; ++b) {
    float dot = 0.0f;
    const float* xb = xm + b * Ee;
    const float* wr = w1 + t * Ee;
    for (int e = 0; e < Ee; ++e) dot += (xb[e] * (1.0f / (float)Ss)) * wr[e];
    float si = dot / (1.0f + __expf(-dot));   // silu
    red[t] = si * w2[t];
    __syncthreads();
    for (int off = 64; off > 0; off >>= 1) {
      if (t < off) red[t] += red[t + off];
      __syncthreads();
    }
    if (t == 0) {
      float learned = 1.0f / (1.0f + __expf(-red[0]));
      float n = (float)Ss * (float)Ee;
      float mu = bstat[2 * b] / n;
      float var = (bstat[2 * b + 1] - n * mu * mu) / (n - 1.0f);
      float vn = 1.0f / (1.0f + __expf(-(var * 10.0f - 5.0f)));
      accs += 0.5f * (vn + learned);
    }
    __syncthreads();
  }
  if (t == 0) {
    float wsf = 256.0f + (accs * 0.25f) * 768.0f;
    int W = (int)wsf;
    if (W > Ss) W = Ss;
    if (W < 256) W = 256;
    *wsout = W;
  }
}

// ---------------------------------------------------------------------------
// Generic bf16 WMMA GEMM: C[M,N](f32) = A[M,K](bf16) * W[N,K]^T(bf16)
// Block tile 128x128, 8 waves, each wave 32(M)x64(N) = 2x4 WMMA tiles.
// K-tiles staged with CDNA5 async global->LDS loads, double-buffered so the
// next tile's fill overlaps the current tile's fragment reads + WMMAs.
// LDS rows padded to 80B to avoid bank conflicts on b128 fragment reads.
// ---------------------------------------------------------------------------
__global__ __launch_bounds__(256) void gemm_bf16_f32(
    const __bf16* __restrict__ A, const __bf16* __restrict__ Bw,
    float* __restrict__ C, int N, int K) {
  __shared__ __align__(16) char lds[2 * 2 * 128 * LDSTR];  // [buf][A|B]
  const int tid = threadIdx.x;
  const int wave = tid >> 5, lane = tid & 31;
  const int l16 = lane & 15, hi = lane >> 4;
  const int bm = blockIdx.x * 128, bn = blockIdx.y * 128;
  const int wm = (wave >> 1) * 32, wn = (wave & 1) * 64;

  auto issue = [&](int k0, int buf) {
    char* bA = lds + buf * (2 * 128 * LDSTR);
    char* bB = bA + 128 * LDSTR;
#pragma unroll
    for (int i = 0; i < 2; ++i) {  // 512 16B chunks per matrix, 256 threads
      int chunk = tid + 256 * i;
      int row = chunk >> 2, c16 = chunk & 3;
      async_load_b128(bA + row * LDSTR + c16 * 16,
                      A + (size_t)(bm + row) * K + k0 + c16 * 8);
      async_load_b128(bB + row * LDSTR + c16 * 16,
                      Bw + (size_t)(bn + row) * K + k0 + c16 * 8);
    }
  };

  f32x8 acc[2][4] = {};

  issue(0, 0);
  for (int k0 = 0; k0 < K; k0 += 32) {
    const int buf = (k0 >> 5) & 1;
    wait_async0();       // my async writes to LDS are complete
    __syncthreads();     // everyone's are -> tile[buf] is ready, tile[buf^1] free
    if (k0 + 32 < K) issue(k0 + 32, buf ^ 1);

    char* ldsA = lds + buf * (2 * 128 * LDSTR);
    char* ldsB = ldsA + 128 * LDSTR;

    bf16x16 af[2], bf[4];
#pragma unroll
    for (int mi = 0; mi < 2; ++mi) {
      // A layout (16-bit 16x32): vg0-3 hold k = hi*8 + 0..7, vg4-7 k = 16+hi*8 + 0..7
      const char* p = ldsA + (wm + mi * 16 + l16) * LDSTR;
      af[mi] = cat16(*(const bf16x8*)(p + hi * 16),
                     *(const bf16x8*)(p + 32 + hi * 16));
    }
#pragma unroll
    for (int ni = 0; ni < 4; ++ni) {
      // B layout (16-bit 32x16): vg0-3 hold k = hi*8 + 0..7, vg4-7 k = 16+hi*8 + 0..7
      const char* p = ldsB + (wn + ni * 16 + l16) * LDSTR + hi * 16;
      bf[ni] = cat16(*(const bf16x8*)p, *(const bf16x8*)(p + 32));
    }
#pragma unroll
    for (int mi = 0; mi < 2; ++mi)
#pragma unroll
      for (int ni = 0; ni < 4; ++ni)
        acc[mi][ni] = wmma_bf16(af[mi], bf[ni], acc[mi][ni]);
  }

#pragma unroll
  for (int mi = 0; mi < 2; ++mi)
#pragma unroll
    for (int ni = 0; ni < 4; ++ni)
#pragma unroll
      for (int r = 0; r < 8; ++r) {
        int row = bm + wm + mi * 16 + r + 8 * hi;   // C/D layout: M = r + 8*hi
        int col = bn + wn + ni * 16 + l16;          // N = lane%16
        C[(size_t)row * N + col] = acc[mi][ni][r];
      }
}

// ---------------------------------------------------------------------------
// Fused dual GEMM with SwiGLU epilogue: out = silu(A*G^T) * (A*U^T), bf16 out
// Same async double-buffered staging, three tiles (A, G, U) per buffer.
// ---------------------------------------------------------------------------
__global__ __launch_bounds__(256) void gemm_dual_silu(
    const __bf16* __restrict__ A, const __bf16* __restrict__ G,
    const __bf16* __restrict__ U, __bf16* __restrict__ out, int N, int K) {
  __shared__ __align__(16) char lds[2 * 3 * 128 * LDSTR];  // [buf][A|G|U]
  const int tid = threadIdx.x;
  const int wave = tid >> 5, lane = tid & 31;
  const int l16 = lane & 15, hi = lane >> 4;
  const int bm = blockIdx.x * 128, bn = blockIdx.y * 128;
  const int wm = (wave >> 1) * 32, wn = (wave & 1) * 64;

  auto issue = [&](int k0, int buf) {
    char* bA = lds + buf * (3 * 128 * LDSTR);
    char* bG = bA + 128 * LDSTR;
    char* bU = bG + 128 * LDSTR;
#pragma unroll
    for (int i = 0; i < 2; ++i) {
      int chunk = tid + 256 * i;
      int row = chunk >> 2, c16 = chunk & 3;
      async_load_b128(bA + row * LDSTR + c16 * 16,
                      A + (size_t)(bm + row) * K + k0 + c16 * 8);
      async_load_b128(bG + row * LDSTR + c16 * 16,
                      G + (size_t)(bn + row) * K + k0 + c16 * 8);
      async_load_b128(bU + row * LDSTR + c16 * 16,
                      U + (size_t)(bn + row) * K + k0 + c16 * 8);
    }
  };

  f32x8 accg[2][4] = {}, accu[2][4] = {};

  issue(0, 0);
  for (int k0 = 0; k0 < K; k0 += 32) {
    const int buf = (k0 >> 5) & 1;
    wait_async0();
    __syncthreads();
    if (k0 + 32 < K) issue(k0 + 32, buf ^ 1);

    char* ldsA = lds + buf * (3 * 128 * LDSTR);
    char* ldsG = ldsA + 128 * LDSTR;
    char* ldsU = ldsG + 128 * LDSTR;

    bf16x16 af[2], gf[4], uf[4];
#pragma unroll
    for (int mi = 0; mi < 2; ++mi) {
      const char* p = ldsA + (wm + mi * 16 + l16) * LDSTR;
      af[mi] = cat16(*(const bf16x8*)(p + hi * 16),
                     *(const bf16x8*)(p + 32 + hi * 16));
    }
#pragma unroll
    for (int ni = 0; ni < 4; ++ni) {
      const char* pg = ldsG + (wn + ni * 16 + l16) * LDSTR + hi * 16;
      const char* pu = ldsU + (wn + ni * 16 + l16) * LDSTR + hi * 16;
      gf[ni] = cat16(*(const bf16x8*)pg, *(const bf16x8*)(pg + 32));
      uf[ni] = cat16(*(const bf16x8*)pu, *(const bf16x8*)(pu + 32));
    }
#pragma unroll
    for (int mi = 0; mi < 2; ++mi)
#pragma unroll
      for (int ni = 0; ni < 4; ++ni) {
        accg[mi][ni] = wmma_bf16(af[mi], gf[ni], accg[mi][ni]);
        accu[mi][ni] = wmma_bf16(af[mi], uf[ni], accu[mi][ni]);
      }
  }

#pragma unroll
  for (int mi = 0; mi < 2; ++mi)
#pragma unroll
    for (int ni = 0; ni < 4; ++ni)
#pragma unroll
      for (int r = 0; r < 8; ++r) {
        int row = bm + wm + mi * 16 + r + 8 * hi;
        int col = bn + wn + ni * 16 + l16;
        float g = accg[mi][ni][r], u = accu[mi][ni][r];
        float s = g / (1.0f + __expf(-g));
        out[(size_t)row * N + col] = (__bf16)(s * u);
      }
}

// ---------------------------------------------------------------------------
// 5) RoPE + scale + bf16 cast. Writes Q,K as [b,h,s,d]; V transposed [b,h,d,s]
//    so attention B-fragments are contiguous 128-bit global loads.
// ---------------------------------------------------------------------------
__global__ __launch_bounds__(256) void rope_kernel(
    const float* __restrict__ qkv, __bf16* __restrict__ qb,
    __bf16* __restrict__ kb, __bf16* __restrict__ vt) {
  const int idx = blockIdx.x * 256 + threadIdx.x;  // over B*H*S*D
  const int d  = idx & (Dd - 1);
  const int s  = (idx >> 6) & (Ss - 1);
  const int bh = idx >> 17;
  const int b = bh >> 3, hh = bh & 7;
  const size_t rb = ((size_t)(b * Ss + s)) * QKVN + hh * Dd;
  float q = qkv[rb + d];
  float k = qkv[rb + Ee + d];
  float v = qkv[rb + 2 * Ee + d];
  const int pr = d ^ 1;
  float qp = qkv[rb + pr];
  float kp = qkv[rb + Ee + pr];
  float rq = (d & 1) ? qp : -qp;   // rotate_half
  float rk = (d & 1) ? kp : -kp;
  float theta = (float)s * __powf(10000.0f, -(float)(d & 31) * (1.0f / 32.0f));
  float sn, cs;
  __sincosf(theta, &sn, &cs);
  float qo = q * cs + rq * sn;
  float ko = k * cs + rk * sn;
  const size_t o = ((size_t)bh * Ss + s) * Dd + d;
  qb[o] = (__bf16)(qo * 0.125f);   // fold scale = D^-0.5 into Q
  kb[o] = (__bf16)ko;
  vt[((size_t)bh * Dd + d) * Ss + s] = (__bf16)v;
}

// ---------------------------------------------------------------------------
// 6) Wave-level flash attention with dynamic causal window.
//    One wave per (b,h,16-query tile); 32-key steps; WMMA for QK^T and PV.
// ---------------------------------------------------------------------------
__global__ __launch_bounds__(256) void flash_kernel(
    const __bf16* __restrict__ qb, const __bf16* __restrict__ kb,
    const __bf16* __restrict__ vt, __bf16* __restrict__ attn,
    const int* __restrict__ wsz) {
  __shared__ __align__(16) char ldsp[8 * 16 * LDSTR];  // per-wave P staging
  const int tid = threadIdx.x, wave = tid >> 5, lane = tid & 31;
  const int l16 = lane & 15, hi = lane >> 4;
  char* P = ldsp + wave * (16 * LDSTR);
  const int W = wsz[0];
  const int gw = blockIdx.x * 8 + wave;      // 0 .. B*H*(S/16)-1
  const int qt = gw & 127, bh = gw >> 7;
  const int qbase = qt * 16;
  const __bf16* qp = qb + (size_t)bh * Ss * Dd;
  const __bf16* kp = kb + (size_t)bh * Ss * Dd;
  const __bf16* vp = vt + (size_t)bh * Dd * Ss;

  // Q A-fragments (two 16x32 chunks over D=64)
  bf16x16 qf[2];
#pragma unroll
  for (int dc = 0; dc < 2; ++dc) {
    const __bf16* p = qp + (size_t)(qbase + l16) * Dd + dc * 32 + hi * 8;
    qf[dc] = cat16(*(const bf16x8*)p, *(const bf16x8*)(p + 16));
  }

  f32x8 o[4] = {};
  float m[8], l[8];
#pragma unroll
  for (int r = 0; r < 8; ++r) { m[r] = -1e30f; l[r] = 0.0f; }

  int kstart = qbase - W + 1;
  if (kstart < 0) kstart = 0;
  kstart &= ~31;
  const int kend = qbase + 16;

  for (int kt = kstart; kt < kend; kt += 32) {
    // scores tile 16q x 32k
    f32x8 sc[2] = {};
#pragma unroll
    for (int nt = 0; nt < 2; ++nt) {
      int key = kt + nt * 16 + l16;
      if (key > Ss - 1) key = Ss - 1;  // masked below; keep loads in-range
      const __bf16* kr = kp + (size_t)key * Dd + hi * 8;
#pragma unroll
      for (int dc = 0; dc < 2; ++dc) {
        bf16x16 bf = cat16(*(const bf16x8*)(kr + dc * 32),
                           *(const bf16x8*)(kr + dc * 32 + 16));
        sc[nt] = wmma_bf16(qf[dc], bf, sc[nt]);
      }
    }

    // mask + online softmax (rows live in accumulator element r, 16 cols/lane-group)
#pragma unroll
    for (int r = 0; r < 8; ++r) {
      const int q = qbase + r + 8 * hi;
      const int k0i = kt + l16, k1i = kt + 16 + l16;
      float s0 = sc[0][r], s1 = sc[1][r];
      if (k0i > q || (q - k0i) >= W) s0 = -1e30f;
      if (k1i > q || (q - k1i) >= W) s1 = -1e30f;
      float rm = fmaxf(s0, s1);
      rm = fmaxf(rm, __shfl_xor(rm, 1, 32));
      rm = fmaxf(rm, __shfl_xor(rm, 2, 32));
      rm = fmaxf(rm, __shfl_xor(rm, 4, 32));
      rm = fmaxf(rm, __shfl_xor(rm, 8, 32));
      float mn = fmaxf(m[r], rm);
      float alpha = __expf(m[r] - mn);
      float p0 = (s0 > -9.0e29f) ? __expf(s0 - mn) : 0.0f;
      float p1 = (s1 > -9.0e29f) ? __expf(s1 - mn) : 0.0f;
      float rs = p0 + p1;
      rs += __shfl_xor(rs, 1, 32);
      rs += __shfl_xor(rs, 2, 32);
      rs += __shfl_xor(rs, 4, 32);
      rs += __shfl_xor(rs, 8, 32);
      l[r] = l[r] * alpha + rs;
      m[r] = mn;
#pragma unroll
      for (int ni = 0; ni < 4; ++ni) o[ni][r] *= alpha;
      __bf16* prow = (__bf16*)(P + (r + 8 * hi) * LDSTR);
      prow[l16]      = (__bf16)p0;
      prow[l16 + 16] = (__bf16)p1;
    }

    // P (16x32) as A-fragment from LDS (same-wave DS ops are in-order)
    const char* pp = P + l16 * LDSTR + hi * 16;
    bf16x16 pf = cat16(*(const bf16x8*)pp, *(const bf16x8*)(pp + 32));
    // V B-fragments straight from transposed global V
#pragma unroll
    for (int nt = 0; nt < 4; ++nt) {
      const __bf16* vr = vp + (size_t)(nt * 16 + l16) * Ss + kt + hi * 8;
      bf16x16 vf = cat16(*(const bf16x8*)vr, *(const bf16x8*)(vr + 16));
      o[nt] = wmma_bf16(pf, vf, o[nt]);
    }
  }

  const int b = bh >> 3, hh = bh & 7;
#pragma unroll
  for (int nt = 0; nt < 4; ++nt)
#pragma unroll
    for (int r = 0; r < 8; ++r) {
      int srow = qbase + r + 8 * hi;
      float v = o[nt][r] / l[r];
      attn[((size_t)(b * Ss + srow)) * Ee + hh * Dd + nt * 16 + l16] = (__bf16)v;
    }
}

// ---------------------------------------------------------------------------
// 8) residual add + RMSNorm2 -> x2 (f32) and h2 (bf16)
// ---------------------------------------------------------------------------
__global__ __launch_bounds__(256) void addnorm_kernel(
    const float* __restrict__ x, const float* __restrict__ proj,
    const float* __restrict__ w, float* __restrict__ x2, __bf16* __restrict__ h2) {
  __shared__ float red[256];
  const int tid = threadIdx.x;
  const int row = blockIdx.x;
  const size_t base = (size_t)row * Ee;
  float v0 = x[base + tid] + proj[base + tid];
  float v1 = x[base + tid + 256] + proj[base + tid + 256];
  x2[base + tid] = v0;
  x2[base + tid + 256] = v1;
  float total = block_sum(v0 * v0 + v1 * v1, red);
  float rms = rsqrtf(total * (1.0f / (float)Ee) + 1e-6f);
  h2[base + tid] = (__bf16)(v0 * rms * w[tid]);
  h2[base + tid + 256] = (__bf16)(v1 * rms * w[tid + 256]);
}

// ---------------------------------------------------------------------------
// Launcher
// ---------------------------------------------------------------------------
extern "C" void kernel_launch(void* const* d_in, const int* in_sizes, int n_in,
                              void* d_out, int out_size, void* d_ws, size_t ws_size,
                              hipStream_t stream) {
  (void)in_sizes; (void)n_in; (void)out_size; (void)ws_size;
  const float* x      = (const float*)d_in[0];
  const float* rms1_w = (const float*)d_in[1];
  const float* rms2_w = (const float*)d_in[2];
  const float* qkv_w  = (const float*)d_in[3];
  const float* out_w  = (const float*)d_in[4];
  const float* cs_w1  = (const float*)d_in[5];
  const float* cs_w2  = (const float*)d_in[6];
  const float* gate_w = (const float*)d_in[7];
  const float* up_w   = (const float*)d_in[8];
  const float* down_w = (const float*)d_in[9];
  float* out = (float*)d_out;

  char* base = (char*)d_ws;
  size_t off = 0;
  auto alloc = [&](size_t bytes) {
    char* p = base + off;
    off += (bytes + 255) & ~(size_t)255;
    return p;
  };

  __bf16* h_bf    = (__bf16*)alloc((size_t)MROWS * Ee * 2);
  float*  qkv_f   = (float*) alloc((size_t)MROWS * QKVN * 4);
  __bf16* qb      = (__bf16*)alloc((size_t)Bb * Hh * Ss * Dd * 2);
  __bf16* kb      = (__bf16*)alloc((size_t)Bb * Hh * Ss * Dd * 2);
  __bf16* vt      = (__bf16*)alloc((size_t)Bb * Hh * Ss * Dd * 2 + 256); // +slack for frag overread
  __bf16* attn    = (__bf16*)alloc((size_t)MROWS * Ee * 2);
  float*  proj    = (float*) alloc((size_t)MROWS * Ee * 4);
  float*  x2      = (float*) alloc((size_t)MROWS * Ee * 4);
  __bf16* h2      = (__bf16*)alloc((size_t)MROWS * Ee * 2);
  __bf16* act     = (__bf16*)alloc((size_t)MROWS * FFN * 2);
  float*  ffn     = (float*) alloc((size_t)MROWS * Ee * 4);
  __bf16* qkvw_b  = (__bf16*)alloc((size_t)QKVN * Ee * 2);
  __bf16* outw_b  = (__bf16*)alloc((size_t)Ee * Ee * 2);
  __bf16* gatew_b = (__bf16*)alloc((size_t)FFN * Ee * 2);
  __bf16* upw_b   = (__bf16*)alloc((size_t)FFN * Ee * 2);
  __bf16* downw_b = (__bf16*)alloc((size_t)Ee * FFN * 2);
  float*  xm      = (float*) alloc((size_t)Bb * Ee * 4);  // 8KB, 256-aligned
  float*  bstat   = (float*) alloc(8 * 4);                // contiguous after xm
  int*    wsi     = (int*)   alloc(4);

  const int NELEM = MROWS * Ee;  // 4194304

  // stats init (xm + bstat are contiguous: 2048 + 8 floats)
  zero_f_kernel<<<9, 256, 0, stream>>>(xm, Bb * Ee + 8);

  // 1) rmsnorm1 + stats
  rmsnorm1_kernel<<<MROWS, 256, 0, stream>>>(x, rms1_w, h_bf, bstat, xm);

  // 2) dynamic window
  cswin_kernel<<<1, 128, 0, stream>>>(bstat, xm, cs_w1, cs_w2, wsi);

  // 3) weight casts
  cast_bf_kernel<<<(QKVN * Ee + 255) / 256, 256, 0, stream>>>(qkv_w,  qkvw_b,  QKVN * Ee);
  cast_bf_kernel<<<(Ee * Ee + 255) / 256,   256, 0, stream>>>(out_w,  outw_b,  Ee * Ee);
  cast_bf_kernel<<<(FFN * Ee + 255) / 256,  256, 0, stream>>>(gate_w, gatew_b, FFN * Ee);
  cast_bf_kernel<<<(FFN * Ee + 255) / 256,  256, 0, stream>>>(up_w,   upw_b,   FFN * Ee);
  cast_bf_kernel<<<(Ee * FFN + 255) / 256,  256, 0, stream>>>(down_w, downw_b, Ee * FFN);

  // 4) QKV GEMM: (8192x512) x (1536x512)^T
  gemm_bf16_f32<<<dim3(MROWS / 128, QKVN / 128), 256, 0, stream>>>(h_bf, qkvw_b, qkv_f, QKVN, Ee);

  // 5) RoPE + layout
  rope_kernel<<<(Bb * Hh * Ss * Dd) / 256, 256, 0, stream>>>(qkv_f, qb, kb, vt);

  // 6) flash attention: 4096 waves, 8 waves/block
  flash_kernel<<<(Bb * Hh * (Ss / 16)) / 8, 256, 0, stream>>>(qb, kb, vt, attn, wsi);

  // 7) out projection
  gemm_bf16_f32<<<dim3(MROWS / 128, Ee / 128), 256, 0, stream>>>(attn, outw_b, proj, Ee, Ee);

  // 8) residual + rmsnorm2
  addnorm_kernel<<<MROWS, 256, 0, stream>>>(x, proj, rms2_w, x2, h2);

  // 9) fused gate/up + SwiGLU
  gemm_dual_silu<<<dim3(MROWS / 128, FFN / 128), 256, 0, stream>>>(h2, gatew_b, upw_b, act, FFN, Ee);

  // 10) down projection
  gemm_bf16_f32<<<dim3(MROWS / 128, Ee / 128), 256, 0, stream>>>(act, downw_b, ffn, Ee, FFN);

  // 11) final residual
  finaladd_kernel<<<NELEM / 256, 256, 0, stream>>>(x2, ffn, out, NELEM);
}
